// AttentionHead_19267223290409
// MI455X (gfx1250) — matrix operand order repaired and checked by reading, not compiled, not run on hardware
//
#include <hip/hip_runtime.h>

typedef __attribute__((ext_vector_type(16))) _Float16 v16h;
typedef __attribute__((ext_vector_type(8)))  _Float16 v8h;
typedef __attribute__((ext_vector_type(4)))  _Float16 v4h;
typedef __attribute__((ext_vector_type(8)))  float    v8f;
typedef __attribute__((ext_vector_type(4)))  float    v4f;

static constexpr int Bn   = 8;
static constexpr int Tn   = 2048;
static constexpr int Dn   = 1024;
static constexpr int HDn  = 64;
static constexpr int NTOK = Bn * Tn;   // 16384

__device__ __forceinline__ v16h cat16(v8h lo, v8h hi) {
  v16h r;
#pragma unroll
  for (int i = 0; i < 8; ++i) { r[i] = lo[i]; r[8 + i] = hi[i]; }
  return r;
}

__device__ __forceinline__ v8f wmma_f16(v16h a, v16h b, v8f c) {
  return __builtin_amdgcn_wmma_f32_16x16x32_f16(
      /*neg_a=*/false, a, /*neg_b=*/false, b,
      /*c_mod=*/(short)0, c, /*reuse_a=*/false, /*reuse_b=*/false);
}

// ---------------------------------------------------------------------------
// Kernel 0: one-shot f32 -> f16 conversion of the three weight matrices so the
// GEMM hot loop has no cvt chain on the load->WMMA path.
// ---------------------------------------------------------------------------
__global__ __launch_bounds__(256) void wcvt_f16(
    const float* __restrict__ wq, const float* __restrict__ wk,
    const float* __restrict__ wv,
    _Float16* __restrict__ hq, _Float16* __restrict__ hk,
    _Float16* __restrict__ hv)
{
  const int i = (blockIdx.x * 256 + threadIdx.x) * 4;   // 64*1024 elems each
  v4f q = *(const v4f*)(wq + i);
  v4f k = *(const v4f*)(wk + i);
  v4f v = *(const v4f*)(wv + i);
  v4h oq, ok, ov;
#pragma unroll
  for (int j = 0; j < 4; ++j) {
    oq[j] = (_Float16)q[j]; ok[j] = (_Float16)k[j]; ov[j] = (_Float16)v[j];
  }
  *(v4h*)(hq + i) = oq;
  *(v4h*)(hk + i) = ok;
  *(v4h*)(hv + i) = ov;
}

// ---------------------------------------------------------------------------
// Kernel 1: q/k/v = x @ w^T + b (f16 WMMA, f32 accum). One wave = one 16-token
// tile computing all 12 (q,k,v x 4 h-tile) outputs -> x streamed from HBM
// exactly once. Per K-step: burst-load ALL 12 B fragments (24 b128 loads) into
// distinct registers, then run the 12 WMMAs back-to-back; launch_bounds(.,1)
// frees the VGPR budget (~210 live) so the scheduler can keep the next
// iteration's loads in flight under the WMMA burst instead of load->wait(0)->
// wmma ping-pong. q gets 1/sqrt(64) folded in; v stored transposed vT[h][tok].
// ---------------------------------------------------------------------------
__global__ __launch_bounds__(128, 1) void qkv_proj_f16_wmma(
    const float* __restrict__ x,
    const _Float16* __restrict__ whq, const _Float16* __restrict__ whk,
    const _Float16* __restrict__ whv,
    const float* __restrict__ bq, const float* __restrict__ bk,
    const float* __restrict__ bv,
    _Float16* __restrict__ qh, _Float16* __restrict__ kh,
    _Float16* __restrict__ vT)
{
  const int lane = threadIdx.x & 31;
  const int wave = threadIdx.x >> 5;
  const int hl   = lane >> 4;          // half-wave 0/1
  const int l16  = lane & 15;
  const int kbA  = hl * 16;            // A-fragment K sub-offset
  const int kbB  = hl * 16;            // B-fragment K sub-offset
  const int m_base = blockIdx.x * 64 + wave * 16;
  const int arow   = m_base + l16;     // token row this lane feeds into A

  const _Float16* wp[3] = {whq, whk, whv};
  v8f acc[3][4] = {};

  for (int kk = 0; kk < Dn; kk += 32) {
    // A fragment: x[arow][kk + kbA .. kbA+15] (f32 -> f16)
    const float* xp = x + (size_t)arow * Dn + kk;
    v4f a0 = *(const v4f*)(xp + kbA);
    v4f a1 = *(const v4f*)(xp + kbA + 4);
    v4f a2 = *(const v4f*)(xp + kbA + 8);
    v4f a3 = *(const v4f*)(xp + kbA + 12);
    v16h a;
#pragma unroll
    for (int i = 0; i < 4; ++i) {
      a[i]      = (_Float16)a0[i];
      a[4 + i]  = (_Float16)a1[i];
      a[8 + i]  = (_Float16)a2[i];
      a[12 + i] = (_Float16)a3[i];
    }

    // Burst-load all 12 B fragments (distinct regs -> one big load burst).
    // B element: w[col][kk + kbB + i], col = n*16 + l16 (B = w^T slice).
    v8h fa[3][4], fb[3][4];
#pragma unroll
    for (int o = 0; o < 3; ++o) {
      const _Float16* w0 = wp[o] + (size_t)l16 * Dn + kk + kbB;
#pragma unroll
      for (int n = 0; n < 4; ++n) {
        const _Float16* w = w0 + (size_t)(n * 16) * Dn;
        fa[o][n] = *(const v8h*)(w);
        fb[o][n] = *(const v8h*)(w + 8);
      }
    }
    // WMMA burst (back-to-back XDL issue).
#pragma unroll
    for (int o = 0; o < 3; ++o)
#pragma unroll
      for (int n = 0; n < 4; ++n)
        acc[o][n] = wmma_f16(a, cat16(fa[o][n], fb[o][n]), acc[o][n]);
  }

  // C/D layout: row = m_base + v + hl*8, col = n*16 + l16
  const int tokbase = m_base + hl * 8;
#pragma unroll
  for (int n = 0; n < 4; ++n) {
    const int h = n * 16 + l16;
    const float biasq = bq[h], biask = bk[h], biasv = bv[h];
    v8h vpack;
#pragma unroll
    for (int v = 0; v < 8; ++v) {
      const int tok = tokbase + v;
      qh[(size_t)tok * HDn + h] = (_Float16)((acc[0][n][v] + biasq) * 0.125f);
      kh[(size_t)tok * HDn + h] = (_Float16)(acc[1][n][v] + biask);
      vpack[v] = (_Float16)(acc[2][n][v] + biasv);
    }
    *(v8h*)(vT + (size_t)h * NTOK + tokbase) = vpack;  // 8 consecutive tokens
  }
}

// ---------------------------------------------------------------------------
// Kernel 2: flash attention, one wave per 16-query tile, 32-key inner tiles.
// V-fragment loads issued at the top of the iteration so their latency hides
// under the S WMMAs + softmax VALU work. P re-swizzled through per-wave LDS.
// ---------------------------------------------------------------------------
__global__ __launch_bounds__(128, 2) void flash_attn_wmma(
    const _Float16* __restrict__ qh, const _Float16* __restrict__ kh,
    const _Float16* __restrict__ vT, float* __restrict__ out)
{
  __shared__ __align__(16) _Float16 p_lds[4][16][40];  // 40-half row stride

  const int lane = threadIdx.x & 31;
  const int wave = threadIdx.x >> 5;
  const int hl   = lane >> 4;
  const int l16  = lane & 15;
  const int kbA  = hl * 16;
  const int kbB  = hl * 16;

  const int batch  = blockIdx.x >> 5;                       // 32 blocks/batch
  const int q_base = (blockIdx.x & 31) * 64 + wave * 16;    // within batch
  const size_t tok0 = (size_t)batch * Tn;

  // Q A-fragments for the two K(=head-dim) steps of 32 (scale pre-folded)
  const _Float16* qp = qh + (tok0 + q_base + l16) * HDn;
  v16h aq0 = cat16(*(const v8h*)(qp + kbA),      *(const v8h*)(qp + kbA + 8));
  v16h aq1 = cat16(*(const v8h*)(qp + 32 + kbA), *(const v8h*)(qp + 40 + kbA));

  v8f acc[4] = {};
  float mrow[8], lrow[8];
#pragma unroll
  for (int v = 0; v < 8; ++v) { mrow[v] = -1e30f; lrow[v] = 0.f; }

  const int ntiles = (q_base + 16 + 31) >> 5;   // causal key range, per wave
  for (int kt = 0; kt < ntiles; ++kt) {
    const int key_base = kt * 32;

    // ---- issue V B-fragment loads early (consumed after softmax) ----
    v8h vfa[4], vfb[4];
#pragma unroll
    for (int n = 0; n < 4; ++n) {
      const _Float16* vp =
          vT + (size_t)(n * 16 + l16) * NTOK + tok0 + key_base + kbB;
      vfa[n] = *(const v8h*)(vp);
      vfb[n] = *(const v8h*)(vp + 8);
    }

    // ---- K B-fragments, batched, then S = Q*K^T ----
    const _Float16* kp0 = kh + (tok0 + key_base + l16) * HDn;
    const _Float16* kp1 = kp0 + 16 * HDn;
    __builtin_prefetch(kp0 + 32 * HDn, 0, 0);   // next key tile
    __builtin_prefetch(kp1 + 32 * HDn, 0, 0);

    v8h k00 = *(const v8h*)(kp0 + kbB);
    v8h k01 = *(const v8h*)(kp0 + kbB + 8);
    v8h k02 = *(const v8h*)(kp0 + 32 + kbB);
    v8h k03 = *(const v8h*)(kp0 + 32 + kbB + 8);
    v8h k10 = *(const v8h*)(kp1 + kbB);
    v8h k11 = *(const v8h*)(kp1 + kbB + 8);
    v8h k12 = *(const v8h*)(kp1 + 32 + kbB);
    v8h k13 = *(const v8h*)(kp1 + 32 + kbB + 8);

    v8f s0 = {}, s1 = {};
    s0 = wmma_f16(aq0, cat16(k00, k01), s0);
    s0 = wmma_f16(aq1, cat16(k02, k03), s0);
    s1 = wmma_f16(aq0, cat16(k10, k11), s1);
    s1 = wmma_f16(aq1, cat16(k12, k13), s1);

    // ---- online softmax; each C row lives across one half-wave ----
#pragma unroll
    for (int v = 0; v < 8; ++v) {
      const int row_g = q_base + v + hl * 8;
      const int c0 = key_base + l16;
      float e0 = (c0      <= row_g) ? s0[v] : -1e30f;   // causal mask
      float e1 = (c0 + 16 <= row_g) ? s1[v] : -1e30f;
      float t = fmaxf(e0, e1);
#pragma unroll
      for (int off = 8; off >= 1; off >>= 1) t = fmaxf(t, __shfl_xor(t, off, 16));
      const float mnew = fmaxf(mrow[v], t);
      const float corr = __expf(mrow[v] - mnew);
      const float p0 = __expf(e0 - mnew);
      const float p1 = __expf(e1 - mnew);
      float rs = p0 + p1;
#pragma unroll
      for (int off = 8; off >= 1; off >>= 1) rs += __shfl_xor(rs, off, 16);
      lrow[v] = lrow[v] * corr + rs;
      mrow[v] = mnew;
#pragma unroll
      for (int n = 0; n < 4; ++n) acc[n][v] *= corr;
      p_lds[wave][v + hl * 8][l16]      = (_Float16)p0;
      p_lds[wave][v + hl * 8][16 + l16] = (_Float16)p1;
    }

    // Same-wave DS store->load: LDS is in-order per wave; drain DScnt and
    // fence the compiler before the transposed read-back.
    asm volatile("s_wait_dscnt 0" ::: "memory");

    // P as WMMA A-fragment (16x32 f16): row = l16, K swizzle via kbA
    const _Float16* pp = &p_lds[wave][l16][0];
    v16h ap = cat16(*(const v8h*)(pp + kbA), *(const v8h*)(pp + kbA + 8));

    // ---- acc += P * V (V fragments already in flight/registers) ----
#pragma unroll
    for (int n = 0; n < 4; ++n)
      acc[n] = wmma_f16(ap, cat16(vfa[n], vfb[n]), acc[n]);
  }

  // normalize and store fp32 output
#pragma unroll
  for (int n = 0; n < 4; ++n) {
#pragma unroll
    for (int v = 0; v < 8; ++v) {
      const int row = q_base + v + hl * 8;
      out[(tok0 + row) * HDn + n * 16 + l16] = acc[n][v] / lrow[v];
    }
  }
}

extern "C" void kernel_launch(void* const* d_in, const int* in_sizes, int n_in,
                              void* d_out, int out_size, void* d_ws, size_t ws_size,
                              hipStream_t stream) {
  const float* x  = (const float*)d_in[0];
  const float* wq = (const float*)d_in[1];
  const float* bq = (const float*)d_in[2];
  const float* wk = (const float*)d_in[3];
  const float* bk = (const float*)d_in[4];
  const float* wv = (const float*)d_in[5];
  const float* bv = (const float*)d_in[6];
  float* out = (float*)d_out;

  // workspace: q,k (token-major f16), vT (h-major f16), f16 weights = ~6.4 MB
  _Float16* qh  = (_Float16*)d_ws;
  _Float16* kh  = qh  + (size_t)NTOK * HDn;
  _Float16* vT  = kh  + (size_t)NTOK * HDn;
  _Float16* whq = vT  + (size_t)NTOK * HDn;
  _Float16* whk = whq + (size_t)HDn * Dn;
  _Float16* whv = whk + (size_t)HDn * Dn;

  wcvt_f16<<<(HDn * Dn) / (256 * 4), 256, 0, stream>>>(wq, wk, wv,
                                                       whq, whk, whv);
  qkv_proj_f16_wmma<<<NTOK / 64, 128, 0, stream>>>(x, whq, whk, whv,
                                                   bq, bk, bv, qh, kh, vT);
  flash_attn_wmma<<<Bn * (Tn / 64), 128, 0, stream>>>(qh, kh, vT, out);
}
